// OwnGRU_41326175322306
// MI455X (gfx1250) — matrix operand order repaired and checked by reading, not compiled, not run on hardware
//
#include <hip/hip_runtime.h>
#include <hip/hip_bf16.h>

#define SEQ   256
#define USERS 256
#define HD    512
#define NCAT  1536   // 3*HD : [z | r | h] segments

// LDS tile: 128 rows x 512 bf16, padded +16B per 1024B row by the TDM
// (breaks 16-way bank conflicts; pitch 1040 B = 520 bf16)
#define LDSPITCH 520
#define LDSTILE  (128 * LDSPITCH)            // elements per staged matrix
#define LDSBYTES (2 * LDSTILE * 2)           // dynamic shared size (260 KB)

typedef __bf16 bf16;
typedef __attribute__((ext_vector_type(16))) __bf16 v16bf;
typedef __attribute__((ext_vector_type(8)))  __bf16 v8bf;
typedef __attribute__((ext_vector_type(8)))  float  v8f;
typedef __attribute__((ext_vector_type(4)))  unsigned int u32x4;
typedef __attribute__((ext_vector_type(8)))  int i32x8;
typedef __attribute__((ext_vector_type(4)))  int i32x4;

// Build a 16-element bf16 fragment from two contiguous 16-byte chunks.
__device__ __forceinline__ v16bf load_frag16(const bf16* p0, const bf16* p1) {
    v8bf lo = *(const v8bf*)p0;
    v8bf hi = *(const v8bf*)p1;
    v16bf r;
#pragma unroll
    for (int i = 0; i < 8; ++i) { r[i] = lo[i]; r[i + 8] = hi[i]; }
    return r;
}

// TDM: DMA a [rows x 512] bf16 slab (row-major, stride 512) from global into
// LDS at byte offset lds_off, inserting 16B of padding after every 1024B row.
// D# layout per CDNA5 ISA (group0 + group1, 2D tensor).
// 6-arg builtin form (clang-23 / therock-10.0 headers).
__device__ __forceinline__ void tdm_load_2d(const bf16* gsrc, unsigned lds_off,
                                            unsigned rows) {
    unsigned long long ga = (unsigned long long)(uintptr_t)gsrc;
    u32x4 g0;
    g0[0] = 1u;                                            // count=1, no gather
    g0[1] = lds_off;                                       // lds_addr
    g0[2] = (unsigned)(ga & 0xFFFFFFFFull);                // global_addr[31:0]
    g0[3] = (unsigned)((ga >> 32) & 0x01FFFFFFull)         // global_addr[56:32]
            | (2u << 30);                                  // type=2 ("image")
    i32x8 g1;
    g1[0] = (int)0x07D10000u;    // mask=0 | data_size=2B | pad_en |
                                 // pad_interval=256dw(1024B) | pad_amount=4dw(16B)
    g1[1] = (int)(512u << 16);   // tensor_dim0 = 512 (in [63:48])
    g1[2] = (int)(rows << 16);   // tensor_dim0[31:16]=0 | tensor_dim1 = rows
    g1[3] = (int)(512u << 16);   // tensor_dim1[31:16]=0 | tile_dim0 = 512
    g1[4] = (int)rows;           // tile_dim1 = rows | tile_dim2 = 0
    g1[5] = 512;                 // tensor_dim0_stride[31:0]
    g1[6] = 0;                   // stride0[47:32] | stride1[15:0]
    g1[7] = 0;                   // tensor_dim1_stride[47:16]
    i32x4 z4 = {0, 0, 0, 0};
    i32x8 z8 = {0, 0, 0, 0, 0, 0, 0, 0};
    __builtin_amdgcn_tensor_load_to_lds(g0, g1, z4, z4, z8, 0);
}

// ---------------------------------------------------------------------------
// Phase 0: convert x, weights, h0 to bf16; pack W/U as Wcat/Ucat [NCAT][HD]
// ---------------------------------------------------------------------------
__global__ __launch_bounds__(256)
void gru_convert(const float* __restrict__ x,
                 const float* __restrict__ Wz, const float* __restrict__ Wr, const float* __restrict__ Wh,
                 const float* __restrict__ Uz, const float* __restrict__ Ur, const float* __restrict__ Uh,
                 const float* __restrict__ h,
                 bf16* __restrict__ Xbf, bf16* __restrict__ Wcat, bf16* __restrict__ Ucat,
                 bf16* __restrict__ Hbf, float* __restrict__ Hf)
{
    const size_t XN = (size_t)SEQ * USERS * HD;
    const size_t WN = (size_t)NCAT * HD;
    const size_t HN = (size_t)USERS * HD;
    const size_t total = XN + 2 * WN + HN;
    for (size_t i = (size_t)blockIdx.x * blockDim.x + threadIdx.x; i < total;
         i += (size_t)gridDim.x * blockDim.x) {
        if (i < XN) {
            Xbf[i] = (bf16)x[i];
        } else if (i < XN + WN) {
            size_t k = i - XN;
            int row = (int)(k / HD), col = (int)(k % HD);
            const float* src = (row < HD) ? Wz : ((row < 2 * HD) ? Wr : Wh);
            Wcat[k] = (bf16)src[(size_t)(row & (HD - 1)) * HD + col];
        } else if (i < XN + 2 * WN) {
            size_t k = i - XN - WN;
            int row = (int)(k / HD), col = (int)(k % HD);
            const float* src = (row < HD) ? Uz : ((row < 2 * HD) ? Ur : Uh);
            Ucat[k] = (bf16)src[(size_t)(row & (HD - 1)) * HD + col];
        } else {
            size_t k = i - XN - 2 * WN;
            float v = h[k];
            Hf[k]  = v;
            Hbf[k] = (bf16)v;
        }
    }
}

// ---------------------------------------------------------------------------
// Per-step dual GEMM: P1 = x_t @ Wcat^T, P2 = h_prev @ Ucat^T  (bf16 -> f32)
// Block tile 128(M) x 128(N); 8 waves; weight slabs staged to LDS via TDM.
// ---------------------------------------------------------------------------
__global__ __launch_bounds__(256)
void gru_gemm_step(const bf16* __restrict__ Xt,    // [USERS][HD]
                   const bf16* __restrict__ Hb,    // [USERS][HD]
                   const bf16* __restrict__ Wcat,  // [NCAT][HD]
                   const bf16* __restrict__ Ucat,  // [NCAT][HD]
                   float* __restrict__ P1,         // [USERS][NCAT]
                   float* __restrict__ P2)         // [USERS][NCAT]
{
    extern __shared__ bf16 smem[];                 // [2][128][LDSPITCH]
    bf16* ldsW = smem;
    bf16* ldsU = smem + LDSTILE;

    const int lane = threadIdx.x & 31;
    const int wave = threadIdx.x >> 5;
    const int m0   = blockIdx.y * 128 + wave * 16;
    const int n0   = blockIdx.x * 128;
    const int lrow = lane & 15;          // m (A) / n (B,C) within tile
    const int lhi  = lane >> 4;          // lane-group select
    const int a_kb = lhi ? 8 : 0;        // A: K halves at kb..kb+7, kb+16..kb+23
    const int b_kb = lhi ? 8 : 0;        // B: K halves at kb..kb+7, kb+16..kb+23

    // Kick off both weight DMAs from wave 0 (TDM ignores EXEC; one issue/wave).
    if (threadIdx.x < 32) {
        tdm_load_2d(Wcat + (size_t)n0 * HD, (unsigned)(uintptr_t)ldsW, 128u);
        tdm_load_2d(Ucat + (size_t)n0 * HD, (unsigned)(uintptr_t)ldsU, 128u);
        __builtin_amdgcn_s_wait_tensorcnt(1);      // W slab landed (in-order)
    }
    __syncthreads();

    const bf16* arow = Xt + (size_t)(m0 + lrow) * HD;
    const bf16* hrow = Hb + (size_t)(m0 + lrow) * HD;

    v8f acc1[8] = {};
    v8f acc2[8] = {};

    // GEMM 1: x_t @ Wcat^T  (B from LDS; U DMA overlaps this compute)
    for (int k0 = 0; k0 < HD; k0 += 32) {
        if (k0 + 32 < HD) __builtin_prefetch(arow + k0 + 32 + a_kb, 0, 1);
        v16bf a = load_frag16(arow + k0 + a_kb, arow + k0 + a_kb + 16);
        v16bf bfr[8];
#pragma unroll
        for (int j = 0; j < 8; ++j) {
            const bf16* bp = ldsW + (size_t)(j * 16 + lrow) * LDSPITCH + k0 + b_kb;
            bfr[j] = load_frag16(bp, bp + 16);
        }
#pragma unroll
        for (int j = 0; j < 8; ++j)
            acc1[j] = __builtin_amdgcn_wmma_f32_16x16x32_bf16(
                false, a, false, bfr[j], (short)0, acc1[j], false, false);
    }

    // U slab must be resident before GEMM 2.
    if (threadIdx.x < 32) __builtin_amdgcn_s_wait_tensorcnt(0);
    __syncthreads();

    // GEMM 2: h_prev @ Ucat^T
    for (int k0 = 0; k0 < HD; k0 += 32) {
        if (k0 + 32 < HD) __builtin_prefetch(hrow + k0 + 32 + a_kb, 0, 1);
        v16bf a = load_frag16(hrow + k0 + a_kb, hrow + k0 + a_kb + 16);
        v16bf bfr[8];
#pragma unroll
        for (int j = 0; j < 8; ++j) {
            const bf16* bp = ldsU + (size_t)(j * 16 + lrow) * LDSPITCH + k0 + b_kb;
            bfr[j] = load_frag16(bp, bp + 16);
        }
#pragma unroll
        for (int j = 0; j < 8; ++j)
            acc2[j] = __builtin_amdgcn_wmma_f32_16x16x32_bf16(
                false, a, false, bfr[j], (short)0, acc2[j], false, false);
    }

    // Write-out per documented C/D layout: VGPR i -> M = i + 8*lhi, N = lane&15
#pragma unroll
    for (int j = 0; j < 8; ++j) {
        const int n = n0 + j * 16 + lrow;
#pragma unroll
        for (int i = 0; i < 8; ++i) {
            const int m = m0 + i + (lhi ? 8 : 0);
            P1[(size_t)m * NCAT + n] = acc1[j][i];
            P2[(size_t)m * NCAT + n] = acc2[j][i];
        }
    }
}

// ---------------------------------------------------------------------------
// Per-step elementwise combine: gates + state update (f32), refresh bf16 state
// ---------------------------------------------------------------------------
__device__ __forceinline__ float sigm_f(float v) { return 1.f / (1.f + __expf(-v)); }
__device__ __forceinline__ float tanh_f(float v) { return 1.f - 2.f / (__expf(2.f * v) + 1.f); }

__global__ __launch_bounds__(256)
void gru_combine(const float* __restrict__ P1, const float* __restrict__ P2,
                 const float* __restrict__ bz, const float* __restrict__ br,
                 const float* __restrict__ bh,
                 float* __restrict__ Hf, bf16* __restrict__ Hbf,
                 float* __restrict__ out_t)
{
    const int idx = blockIdx.x * blockDim.x + threadIdx.x;   // USERS*HD threads
    const int u = idx >> 9;          // / HD
    const int c = idx & (HD - 1);
    const size_t base = (size_t)u * NCAT + c;

    float z  = sigm_f(P1[base]          + P2[base]          + bz[c]);
    float r  = sigm_f(P1[base + HD]     + P2[base + HD]     + br[c]);
    float ht = tanh_f(P1[base + 2 * HD] + r * P2[base + 2 * HD] + bh[c]);
    float hp = Hf[idx];
    float hn = z * hp + (1.f - z) * ht;

    out_t[idx] = hn;
    Hf[idx]    = hn;
    Hbf[idx]   = (bf16)hn;
}

// ---------------------------------------------------------------------------
// Launch: convert once, then 256 x (gemm + combine), then copy h_last.
// Workspace: Xbf 64MB + Wcat/Ucat 3MB + Hbf 0.25MB + Hf 0.5MB + P1/P2 3MB ~ 71MB
// ---------------------------------------------------------------------------
extern "C" void kernel_launch(void* const* d_in, const int* in_sizes, int n_in,
                              void* d_out, int out_size, void* d_ws, size_t ws_size,
                              hipStream_t stream) {
    const float* x  = (const float*)d_in[0];
    const float* h  = (const float*)d_in[1];
    const float* Wz = (const float*)d_in[2];
    const float* Wr = (const float*)d_in[3];
    const float* Wh = (const float*)d_in[4];
    const float* Uz = (const float*)d_in[5];
    const float* Ur = (const float*)d_in[6];
    const float* Uh = (const float*)d_in[7];
    const float* bz = (const float*)d_in[8];
    const float* br = (const float*)d_in[9];
    const float* bh = (const float*)d_in[10];
    float* out = (float*)d_out;

    char* p = (char*)d_ws;
    bf16*  Xbf  = (bf16*)p;  p += (size_t)SEQ * USERS * HD * sizeof(bf16);
    bf16*  Wcat = (bf16*)p;  p += (size_t)NCAT * HD * sizeof(bf16);
    bf16*  Ucat = (bf16*)p;  p += (size_t)NCAT * HD * sizeof(bf16);
    bf16*  Hbf  = (bf16*)p;  p += (size_t)USERS * HD * sizeof(bf16);
    float* Hf   = (float*)p; p += (size_t)USERS * HD * sizeof(float);
    float* P1   = (float*)p; p += (size_t)USERS * NCAT * sizeof(float);
    float* P2   = (float*)p;

    gru_convert<<<4096, 256, 0, stream>>>(x, Wz, Wr, Wh, Uz, Ur, Uh, h,
                                          Xbf, Wcat, Ucat, Hbf, Hf);

    dim3 ggrid(NCAT / 128, USERS / 128);   // 12 x 2 blocks, 256 threads each
    const int cblocks = (USERS * HD) / 256;
    for (int t = 0; t < SEQ; ++t) {
        gru_gemm_step<<<ggrid, 256, LDSBYTES, stream>>>(
            Xbf + (size_t)t * USERS * HD, Hbf, Wcat, Ucat, P1, P2);
        gru_combine<<<cblocks, 256, 0, stream>>>(P1, P2, bz, br, bh,
                                                 Hf, Hbf,
                                                 out + (size_t)t * USERS * HD);
    }
    // h_last -> tail of output
    (void)hipMemcpyAsync(out + (size_t)SEQ * USERS * HD, Hf,
                         (size_t)USERS * HD * sizeof(float),
                         hipMemcpyDeviceToDevice, stream);
}
